// mLSTM_11922829213849
// MI455X (gfx1250) — matrix operand order repaired and checked by reading, not compiled, not run on hardware
//
#include <hip/hip_runtime.h>

typedef __attribute__((ext_vector_type(16))) _Float16 v16h;
typedef __attribute__((ext_vector_type(8)))  _Float16 v8h;
typedef __attribute__((ext_vector_type(8)))  float    v8f;
typedef __attribute__((ext_vector_type(4)))  int      v4i;

static constexpr int BB = 8, SS = 2048, DD = 1024, HH = 1024, LL = 2;
static constexpr int M_TOT = BB * SS;                 // 16384 GEMM rows
static constexpr size_t MN = (size_t)M_TOT * HH;      // 16,777,216 per gate buffer
static constexpr size_t HD = (size_t)HH * DD;         // 1,048,576 per weight matrix
static constexpr int CC = 32;                         // scan chunks per sequence
static constexpr int CL = SS / CC;                    // 64 steps per chunk
static constexpr size_t SEG = (size_t)BB * CC * HH;   // 262,144 chunk records

// ---------------------------------------------------------------------------
// CDNA5 async global->LDS staging (ASYNCcnt path), guarded so the file still
// compiles on toolchains that do not declare the builtin.
// Probe round 3: builtin exists, signature is
//   (int4 addrspace(1)*, int4 addrspace(3)*, imm int offset, imm int cpol)
// ---------------------------------------------------------------------------
#if defined(__has_builtin)
#  if __has_builtin(__builtin_amdgcn_global_load_async_to_lds_b128)
#    define HAVE_ASYNC_LDS 1
#  endif
#endif
#ifndef HAVE_ASYNC_LDS
#  define HAVE_ASYNC_LDS 0
#endif

#if HAVE_ASYNC_LDS
__device__ __forceinline__ void asyncCopy16(const _Float16* g, _Float16* l) {
  __builtin_amdgcn_global_load_async_to_lds_b128(
      (__attribute__((address_space(1))) v4i*)g,
      (__attribute__((address_space(3))) v4i*)l, 0, 0);
}
__device__ __forceinline__ void asyncWait0() {
#  if __has_builtin(__builtin_amdgcn_s_wait_asynccnt)
  __builtin_amdgcn_s_wait_asynccnt(0);
#  else
  asm volatile("s_wait_asynccnt 0" ::: "memory");
#  endif
}
#endif

// ---------------------------------------------------------------------------
// fp32 -> f16 conversion (x, weights)
// ---------------------------------------------------------------------------
__global__ void cvt_f32_to_f16(const float* __restrict__ src,
                               _Float16* __restrict__ dst, int n) {
  int i = blockIdx.x * 256 + threadIdx.x;
  if (i < n) dst[i] = (_Float16)src[i];
}

// ---------------------------------------------------------------------------
// Fused 6-gate GEMM + bias + activation, double-buffered LDS pipeline.
//   C[m,n] = sum_k X[m,k] * W[n,k]   (X: [M_TOT,D] f16, W: [H,D] f16 per gate)
// Block: 256 threads (8 wave32), tile 128(M) x 128(N), K staged 32 at a time.
// Wave grid 4(M) x 2(N): each wave owns 32x64 -> 2x4 WMMA accumulators.
// grid = (N/128, M/128, 6 gates)
// ---------------------------------------------------------------------------
static constexpr int LDST = 40;   // padded LDS row stride in halves (80 B)

__global__ void __launch_bounds__(256)
gemm_gates(const _Float16* __restrict__ X,
           const _Float16* __restrict__ W6,    // [6, H, D] f16
           const float* __restrict__ bi, const float* __restrict__ bf,
           const float* __restrict__ bo, const float* __restrict__ bq,
           const float* __restrict__ bk, const float* __restrict__ bv,
           float* __restrict__ gates)          // [6, M_TOT, H] f32
{
  __shared__ _Float16 lA[2][128 * LDST];
  __shared__ _Float16 lB[2][128 * LDST];

  const int tid   = threadIdx.x;
  const int gate  = blockIdx.z;
  const int mBase = blockIdx.y * 128;
  const int nBase = blockIdx.x * 128;

  const int lane  = tid & 31;
  const int wave  = tid >> 5;
  const int l15   = lane & 15;
  const int hi    = lane >> 4;
  const int waveM = wave >> 1;   // 0..3
  const int waveN = wave & 1;    // 0..1

  const _Float16* Wg = W6 + (size_t)gate * HD;

  v8f acc[2][4];
  const v8f vzero = {};
#pragma unroll
  for (int mt = 0; mt < 2; ++mt)
#pragma unroll
    for (int nt = 0; nt < 4; ++nt) acc[mt][nt] = vzero;

  // staging: each thread moves two 16B segments of A and of B per K-chunk
  const int r0 = tid >> 2;              // 0..63
  const int p0 = (tid & 3) * 8;         // half offset within row: 0,8,16,24
  const size_t aRow0 = (size_t)(mBase + r0)      * DD + p0;
  const size_t aRow1 = (size_t)(mBase + r0 + 64) * DD + p0;
  const size_t bRow0 = (size_t)(nBase + r0)      * DD + p0;
  const size_t bRow1 = (size_t)(nBase + r0 + 64) * DD + p0;
  const int sA0 = (r0)      * LDST + p0;
  const int sA1 = (r0 + 64) * LDST + p0;

  // prologue: stage chunk 0 into buffer 0
#if HAVE_ASYNC_LDS
  asyncCopy16(X  + aRow0, &lA[0][sA0]);
  asyncCopy16(X  + aRow1, &lA[0][sA1]);
  asyncCopy16(Wg + bRow0, &lB[0][sA0]);
  asyncCopy16(Wg + bRow1, &lB[0][sA1]);
#else
  {
    v8h a0 = *(const v8h*)(X  + aRow0);
    v8h a1 = *(const v8h*)(X  + aRow1);
    v8h b0 = *(const v8h*)(Wg + bRow0);
    v8h b1 = *(const v8h*)(Wg + bRow1);
    *(v8h*)&lA[0][sA0] = a0;  *(v8h*)&lA[0][sA1] = a1;
    *(v8h*)&lB[0][sA0] = b0;  *(v8h*)&lB[0][sA1] = b1;
  }
#endif

  for (int kk = 0; kk < DD / 32; ++kk) {
    const int cur = kk & 1;

#if HAVE_ASYNC_LDS
    asyncWait0();          // my async loads into buffer `cur` have landed
    __syncthreads();       // everyone's landed; buffer cur^1 free to refill
    if (kk < DD / 32 - 1) {
      const int k0 = (kk + 1) * 32;
      asyncCopy16(X  + aRow0 + k0, &lA[cur ^ 1][sA0]);
      asyncCopy16(X  + aRow1 + k0, &lA[cur ^ 1][sA1]);
      asyncCopy16(Wg + bRow0 + k0, &lB[cur ^ 1][sA0]);
      asyncCopy16(Wg + bRow1 + k0, &lB[cur ^ 1][sA1]);
    }
#else
    __syncthreads();
    v8h na0, na1, nb0, nb1;
    if (kk < DD / 32 - 1) {              // fetch next chunk while we compute
      const int k0 = (kk + 1) * 32;
      na0 = *(const v8h*)(X  + aRow0 + k0);
      na1 = *(const v8h*)(X  + aRow1 + k0);
      nb0 = *(const v8h*)(Wg + bRow0 + k0);
      nb1 = *(const v8h*)(Wg + bRow1 + k0);
    }
#endif

    // A fragment: lane(l15,hi) row M=l15; halves j<8 -> K=8*hi+j, j>=8 -> K=16+8*hi+(j-8)
    v16h afrag[2];
#pragma unroll
    for (int mt = 0; mt < 2; ++mt) {
      const int row = waveM * 32 + mt * 16 + l15;
      v8h lo = *(const v8h*)&lA[cur][row * LDST + hi * 8];
      v8h hh = *(const v8h*)&lA[cur][row * LDST + 16 + hi * 8];
      afrag[mt] = __builtin_shufflevector(lo, hh, 0, 1, 2, 3, 4, 5, 6, 7,
                                                  8, 9, 10, 11, 12, 13, 14, 15);
    }
    // B fragment: lane holds column N=l15 of B (= row of W); halves j<8 -> K=8*hi+j, j>=8 -> K=16+8*hi+(j-8)
    v16h bfrag[4];
#pragma unroll
    for (int nt = 0; nt < 4; ++nt) {
      const int row = waveN * 64 + nt * 16 + l15;
      v8h lo = *(const v8h*)&lB[cur][row * LDST + hi * 8];
      v8h hh = *(const v8h*)&lB[cur][row * LDST + 16 + hi * 8];
      bfrag[nt] = __builtin_shufflevector(lo, hh, 0, 1, 2, 3, 4, 5, 6, 7,
                                                  8, 9, 10, 11, 12, 13, 14, 15);
    }

#pragma unroll
    for (int mt = 0; mt < 2; ++mt)
#pragma unroll
      for (int nt = 0; nt < 4; ++nt)
        acc[mt][nt] = __builtin_amdgcn_wmma_f32_16x16x32_f16(
            false, afrag[mt], false, bfrag[nt], (short)0, acc[mt][nt],
            false, false);

#if !HAVE_ASYNC_LDS
    if (kk < DD / 32 - 1) {
      *(v8h*)&lA[cur ^ 1][sA0] = na0;  *(v8h*)&lA[cur ^ 1][sA1] = na1;
      *(v8h*)&lB[cur ^ 1][sA0] = nb0;  *(v8h*)&lB[cur ^ 1][sA1] = nb1;
    }
#endif
  }

  // epilogue: bias + per-gate activation, nontemporal store of fp32 gate buffer
  const float* bias = (gate == 0) ? bi : (gate == 1) ? bf : (gate == 2) ? bo
                    : (gate == 3) ? bq : (gate == 4) ? bk : bv;
  float* out = gates + (size_t)gate * MN;

  float bval[4];
#pragma unroll
  for (int nt = 0; nt < 4; ++nt)
    bval[nt] = bias[nBase + waveN * 64 + nt * 16 + l15];

#pragma unroll
  for (int mt = 0; mt < 2; ++mt) {
#pragma unroll
    for (int nt = 0; nt < 4; ++nt) {
      const int n = nBase + waveN * 64 + nt * 16 + l15;
#pragma unroll
      for (int g = 0; g < 8; ++g) {
        const int m = mBase + waveM * 32 + mt * 16 + g + 8 * hi;  // C layout: M = g + 8*hi
        float v = acc[mt][nt][g] + bval[nt];
        if (gate == 0)      v = __expf(v);                        // i gate
        else if (gate <= 2) v = 1.0f / (1.0f + __expf(-v));       // f, o gates
        else if (gate == 4) v *= 0.03125f;                        // k * 1/sqrt(H)
        __builtin_nontemporal_store(v, &out[(size_t)m * HH + n]);
      }
    }
  }
}

// ---------------------------------------------------------------------------
// Segmented scan, phase 1: per (b, chunk, h) compute chunk summary
//   F = prod f,  Uc/Un = chunk contribution with zero initial state.
// ---------------------------------------------------------------------------
__global__ void __launch_bounds__(256)
scan_phase1(const float* __restrict__ gates,
            float* __restrict__ Farr, float* __restrict__ Ucarr,
            float* __restrict__ Unarr)
{
  const int t    = blockIdx.x * 256 + threadIdx.x;   // 0 .. B*CC*H-1
  const int h    = t & (HH - 1);
  const int rest = t >> 10;                          // b*CC + chunk
  const int chunk = rest & (CC - 1);
  const int b     = rest >> 5;

  const float* gi = gates;
  const float* gf = gates + MN;
  const float* gk = gates + 4 * MN;
  const float* gv = gates + 5 * MN;
  const size_t base = ((size_t)b * SS + chunk * CL) * HH + h;

  float F = 1.0f, Uc = 0.0f, Un = 0.0f;
  for (int u = 0; u < CL; u += 8) {
    if (u + 8 < CL) {
      const size_t pf = base + (size_t)(u + 8) * HH;
      __builtin_prefetch((const void*)(gi + pf), 0, 0);
      __builtin_prefetch((const void*)(gf + pf), 0, 0);
      __builtin_prefetch((const void*)(gk + pf), 0, 0);
      __builtin_prefetch((const void*)(gv + pf), 0, 0);
    }
    float iv[8], fv[8], kv[8], vv[8];
#pragma unroll
    for (int j = 0; j < 8; ++j) {
      const size_t off = base + (size_t)(u + j) * HH;
      iv[j] = __builtin_nontemporal_load(gi + off);
      fv[j] = __builtin_nontemporal_load(gf + off);
      kv[j] = __builtin_nontemporal_load(gk + off);
      vv[j] = __builtin_nontemporal_load(gv + off);
    }
#pragma unroll
    for (int j = 0; j < 8; ++j) {
      F  = F * fv[j];
      Uc = fv[j] * Uc + iv[j] * (vv[j] * kv[j]);
      Un = fv[j] * Un + iv[j] * kv[j];
    }
  }
  Farr[t] = F; Ucarr[t] = Uc; Unarr[t] = Un;   // t == (b*CC+chunk)*H + h
}

// ---------------------------------------------------------------------------
// Phase 2: per-channel scan over 32 chunk summaries -> per-chunk (c0, n0);
// also emits c_last / n_last for this layer.
// ---------------------------------------------------------------------------
__global__ void __launch_bounds__(256)
scan_phase2(const float* __restrict__ Farr, const float* __restrict__ Ucarr,
            const float* __restrict__ Unarr,
            float* __restrict__ c0a, float* __restrict__ n0a,
            float* __restrict__ lastBase, int layer)
{
  const int idx = blockIdx.x * 256 + threadIdx.x;    // b*H + h
  const int b = idx >> 10;
  const int h = idx & (HH - 1);
  float c = 0.0f, n = 0.0f;
#pragma unroll
  for (int ch = 0; ch < CC; ++ch) {
    const int s = (b * CC + ch) * HH + h;
    c0a[s] = c; n0a[s] = n;
    const float F = Farr[s];
    c = F * c + Ucarr[s];
    n = F * n + Unarr[s];
  }
  lastBase[LL * BB * HH + layer * BB * HH + idx] = c;       // c_last
  lastBase[2 * LL * BB * HH + layer * BB * HH + idx] = n;   // n_last
}

// ---------------------------------------------------------------------------
// Phase 3: re-scan each chunk from (c0, n0), emit h stream (+ h_last).
// ---------------------------------------------------------------------------
__global__ void __launch_bounds__(256)
scan_phase3(const float* __restrict__ gates,
            const float* __restrict__ c0a, const float* __restrict__ n0a,
            float* __restrict__ hOutF32,      // non-null for last layer
            _Float16* __restrict__ hOutF16,   // non-null for layer 0
            float* __restrict__ lastBase, int layer)
{
  const int t    = blockIdx.x * 256 + threadIdx.x;
  const int h    = t & (HH - 1);
  const int rest = t >> 10;
  const int chunk = rest & (CC - 1);
  const int b     = rest >> 5;

  const float* gi = gates;
  const float* gf = gates + MN;
  const float* go = gates + 2 * MN;
  const float* gq = gates + 3 * MN;
  const float* gk = gates + 4 * MN;
  const float* gv = gates + 5 * MN;
  const size_t base = ((size_t)b * SS + chunk * CL) * HH + h;

  float c = c0a[t], n = n0a[t], hval = 0.0f;
  for (int u = 0; u < CL; u += 8) {
    if (u + 8 < CL) {
      const size_t pf = base + (size_t)(u + 8) * HH;
      __builtin_prefetch((const void*)(gi + pf), 0, 0);
      __builtin_prefetch((const void*)(gf + pf), 0, 0);
      __builtin_prefetch((const void*)(go + pf), 0, 0);
      __builtin_prefetch((const void*)(gq + pf), 0, 0);
      __builtin_prefetch((const void*)(gk + pf), 0, 0);
      __builtin_prefetch((const void*)(gv + pf), 0, 0);
    }
    float iv[8], fv[8], ov[8], qv[8], kv[8], vv[8];
#pragma unroll
    for (int j = 0; j < 8; ++j) {
      const size_t off = base + (size_t)(u + j) * HH;
      iv[j] = __builtin_nontemporal_load(gi + off);
      fv[j] = __builtin_nontemporal_load(gf + off);
      ov[j] = __builtin_nontemporal_load(go + off);
      qv[j] = __builtin_nontemporal_load(gq + off);
      kv[j] = __builtin_nontemporal_load(gk + off);
      vv[j] = __builtin_nontemporal_load(gv + off);
    }
#pragma unroll
    for (int j = 0; j < 8; ++j) {
      c = fv[j] * c + iv[j] * (vv[j] * kv[j]);
      n = fv[j] * n + iv[j] * kv[j];
      const float dnm = fmaxf(fabsf(n * qv[j]), 1.0f);
      hval = ov[j] * (c * qv[j] / dnm);
      const size_t off = base + (size_t)(u + j) * HH;
      if (hOutF32) __builtin_nontemporal_store(hval, &hOutF32[off]);
      else         hOutF16[off] = (_Float16)hval;   // cached: next layer's GEMM input
    }
  }
  if (chunk == CC - 1)
    lastBase[layer * BB * HH + b * HH + h] = hval;          // h_last
}

// ---------------------------------------------------------------------------
// Host orchestration
// ---------------------------------------------------------------------------
extern "C" void kernel_launch(void* const* d_in, const int* in_sizes, int n_in,
                              void* d_out, int out_size, void* d_ws, size_t ws_size,
                              hipStream_t stream) {
  (void)in_sizes; (void)n_in; (void)out_size; (void)ws_size;

  const float* x = (const float*)d_in[0];
  const float* Wg[6] = {(const float*)d_in[1], (const float*)d_in[2],
                        (const float*)d_in[3], (const float*)d_in[4],
                        (const float*)d_in[5], (const float*)d_in[6]};
  const float* bg[6] = {(const float*)d_in[7], (const float*)d_in[8],
                        (const float*)d_in[9], (const float*)d_in[10],
                        (const float*)d_in[11], (const float*)d_in[12]};
  float* out = (float*)d_out;

  // workspace layout
  char* ws = (char*)d_ws;
  size_t off = 0;
  float*    gates = (float*)(ws + off);    off += 6 * MN * sizeof(float);
  _Float16* xf16  = (_Float16*)(ws + off); off += MN * sizeof(_Float16);   // B*S*D == MN
  _Float16* wf16  = (_Float16*)(ws + off); off += 6 * HD * sizeof(_Float16);
  float*    Farr  = (float*)(ws + off);    off += SEG * sizeof(float);
  float*    Ucarr = (float*)(ws + off);    off += SEG * sizeof(float);
  float*    Unarr = (float*)(ws + off);    off += SEG * sizeof(float);
  float*    c0a   = (float*)(ws + off);    off += SEG * sizeof(float);
  float*    n0a   = (float*)(ws + off);    off += SEG * sizeof(float);

  const dim3 gemmGrid(HH / 128, M_TOT / 128, 6);
  const int segBlocks  = (int)(SEG / 256);       // 1024
  const int chanBlocks = BB * HH / 256;          // 32

  for (int l = 0; l < LL; ++l) {
    if (l == 0)  // layer 0 input: convert x; layer 1 input was written f16 by phase 3
      cvt_f32_to_f16<<<(int)(MN / 256), 256, 0, stream>>>(x, xf16, (int)MN);

    for (int g = 0; g < 6; ++g)
      cvt_f32_to_f16<<<(int)(HD / 256), 256, 0, stream>>>(
          Wg[g] + (size_t)l * HD, wf16 + (size_t)g * HD, (int)HD);

    gemm_gates<<<gemmGrid, 256, 0, stream>>>(
        xf16, wf16,
        bg[0] + l * HH, bg[1] + l * HH, bg[2] + l * HH,
        bg[3] + l * HH, bg[4] + l * HH, bg[5] + l * HH,
        gates);

    scan_phase1<<<segBlocks, 256, 0, stream>>>(gates, Farr, Ucarr, Unarr);
    scan_phase2<<<chanBlocks, 256, 0, stream>>>(Farr, Ucarr, Unarr, c0a, n0a,
                                                out + MN, l);
    if (l == LL - 1)
      scan_phase3<<<segBlocks, 256, 0, stream>>>(gates, c0a, n0a, out,
                                                 (_Float16*)nullptr, out + MN, l);
    else
      scan_phase3<<<segBlocks, 256, 0, stream>>>(gates, c0a, n0a, (float*)nullptr,
                                                 xf16, out + MN, l);
  }
}
